// LTE_38637525795039
// MI455X (gfx1250) — compile-verified, matrix operand order checked by reading, not run on hardware
//
#include <hip/hip_runtime.h>
#include <hip/hip_bf16.h>

typedef _Float16 half_t;
typedef __attribute__((ext_vector_type(16))) _Float16 v16h;
typedef __attribute__((ext_vector_type(8)))  _Float16 v8h;
typedef __attribute__((ext_vector_type(8)))  float    v8f;

// Problem constants (from setup_inputs)
#define HF   128
#define WF   128
#define CIN  64
#define HD   256
#define NOUT 512   // coef(256) || freq(256) from one fused implicit GEMM
#define HO   256
#define WO   256

// Workspace layout (offsets in half_t elements)
#define OFF_FEATH 0u
#define N_FEATH   (2u*HF*WF*CIN)                     // 2,097,152 : feat as [img][y][x][c] f16
#define OFF_CONVW (OFF_FEATH + N_FEATH)
#define N_CONVW   (18u*32u*32u*16u)                  //   294,912 : conv B-frags (K=576,N=512)
#define OFF_MLPW  (OFF_CONVW + N_CONVW)
#define N_MLPW    (3u*8u*16u*32u*16u + 8u*32u*16u)   //   200,704 : mlp w1..w3 + w4 B-frags
#define OFF_CF    (OFF_MLPW + N_MLPW)
#define N_CF      (2u*HF*WF*(unsigned)NOUT)          // 16,777,216 : coef||freq, channel-last f16

#define PI_F 3.14159265358979f

// Async global->LDS path (gfx1250): guarded so compile can never fail.
#if defined(__HIP_DEVICE_COMPILE__) && __has_builtin(__builtin_amdgcn_global_load_async_to_lds_b128)
#define HAVE_ASYNC_LDS 1
// Param 1 of the builtin is: int4(vector_size 16), address_space(1), non-const.
typedef int v4i_gcc __attribute__((vector_size(16)));
typedef __attribute__((address_space(1))) v4i_gcc* g_v4i_p;
typedef __attribute__((address_space(3))) v4i_gcc* l_v4i_p;
#endif
#if defined(__HIP_DEVICE_COMPILE__) && __has_builtin(__builtin_amdgcn_s_wait_asynccnt)
#define WAIT_ASYNC() __builtin_amdgcn_s_wait_asynccnt(0)
#else
#define WAIT_ASYNC() asm volatile("s_wait_asynccnt 0" ::: "memory")
#endif

// ---- small vector helpers -------------------------------------------------
static __device__ inline v8h v8h_zero() {
  v8h r;
#pragma unroll
  for (int i = 0; i < 8; ++i) r[i] = (half_t)0;
  return r;
}
static __device__ inline v16h cat16(v8h lo, v8h hi) {
  return __builtin_shufflevector(lo, hi, 0,1,2,3,4,5,6,7,8,9,10,11,12,13,14,15);
}

// Nearest-gather coordinate math (matches _nearest_idx / rel computation).
static __device__ inline void shift_rel(float cy, float cx, int s,
                                        int& iy, int& ix, float& rel0, float& rel1) {
  const float rx = 1.0f / 128.0f, eps = 1e-6f;
  float vx = (s < 2) ? -1.f : 1.f;          // loop order: (-1,-1),(-1,1),(1,-1),(1,1)
  float vy = (s & 1) ? 1.f : -1.f;
  float cyc = fminf(fmaxf(cy + vx * rx + eps, -1.f + 1e-6f), 1.f - 1e-6f);
  float cxc = fminf(fmaxf(cx + vy * rx + eps, -1.f + 1e-6f), 1.f - 1e-6f);
  float fy = ((cyc + 1.f) * 128.f - 1.f) * 0.5f;
  float fx = ((cxc + 1.f) * 128.f - 1.f) * 0.5f;
  iy = (int)fminf(fmaxf(rintf(fy), 0.f), 127.f);
  ix = (int)fminf(fmaxf(rintf(fx), 0.f), 127.f);
  float qy = (2.f * iy + 1.f) / 128.f - 1.f;
  float qx = (2.f * ix + 1.f) / 128.f - 1.f;
  rel0 = (cy - qy) * 128.f;
  rel1 = (cx - qx) * 128.f;
}

// ---- Kernel 0: pack f32 inputs into f16 / WMMA-fragment layouts ----------
// B-fragment layout convention (16-bit, 32x16 K x N tile), per lane L:
//   n = L%16, kbase = (L<16)?0:8; halves i<8 -> K=kbase+i, i>=8 -> K=kbase+16+(i-8)
// (mirrors the documented 16-bit A layout; per-lane data is 32B contiguous)
__global__ void pack_kernel(const float* __restrict__ feat,
                            const float* __restrict__ coef_w,
                            const float* __restrict__ freq_w,
                            const float* __restrict__ w1,
                            const float* __restrict__ w2,
                            const float* __restrict__ w3,
                            const float* __restrict__ w4,
                            half_t* __restrict__ ws) {
  const unsigned total = N_FEATH + N_CONVW + N_MLPW;
  for (unsigned idx = blockIdx.x * blockDim.x + threadIdx.x; idx < total;
       idx += gridDim.x * blockDim.x) {
    if (idx < N_FEATH) {
      unsigned t = idx;
      unsigned c = t & 63u;  t >>= 6;
      unsigned x = t & 127u; t >>= 7;
      unsigned y = t & 127u; t >>= 7;
      unsigned img = t;
      ws[OFF_FEATH + idx] = (half_t)feat[((img * 64u + c) * 128u + y) * 128u + x];
    } else if (idx < N_FEATH + N_CONVW) {
      unsigned t = idx - N_FEATH;
      unsigned i  = t & 15u;
      unsigned L  = (t >> 4) & 31u;
      unsigned nt = (t >> 9) & 31u;
      unsigned kc = t >> 14;                 // 0..17
      unsigned kbase = (L < 16u) ? 0u : 8u;
      unsigned koff  = (i < 8u) ? (kbase + i) : (kbase + 16u + (i - 8u));
      unsigned K = kc * 32u + koff;          // 0..575
      unsigned n = nt * 16u + (L & 15u);     // 0..511
      unsigned tap = K / 64u, ci = K % 64u;
      unsigned ky = tap / 3u, kx = tap % 3u;
      float v = (n < 256u)
                    ? coef_w[((n * 64u + ci) * 3u + ky) * 3u + kx]
                    : freq_w[(((n - 256u) * 64u + ci) * 3u + ky) * 3u + kx];
      ws[OFF_CONVW + t] = (half_t)v;
    } else {
      unsigned t = idx - (N_FEATH + N_CONVW);
      if (t < 196608u) {                     // layers 0..2 : [8kc][16nt][32L][16i]
        unsigned l   = t / 65536u;
        unsigned rem = t - l * 65536u;
        unsigned kc = rem >> 13;
        unsigned nt = (rem >> 9) & 15u;
        unsigned L  = (rem >> 4) & 31u;
        unsigned i  = rem & 15u;
        unsigned kbase = (L < 16u) ? 0u : 8u;
        unsigned koff  = (i < 8u) ? (kbase + i) : (kbase + 16u + (i - 8u));
        unsigned K = kc * 32u + koff;
        unsigned n = nt * 16u + (L & 15u);
        const float* W = (l == 0) ? w1 : (l == 1) ? w2 : w3;
        ws[OFF_MLPW + t] = (half_t)W[n * 256u + K];   // B[k][n] = W[n][k]
      } else {                               // layer 3 (w4): [8kc][32L][16i], N padded to 16
        unsigned rem = t - 196608u;
        unsigned kc = rem >> 9;
        unsigned L  = (rem >> 4) & 31u;
        unsigned i  = rem & 15u;
        unsigned kbase = (L < 16u) ? 0u : 8u;
        unsigned koff  = (i < 8u) ? (kbase + i) : (kbase + 16u + (i - 8u));
        unsigned K = kc * 32u + koff;
        unsigned n = L & 15u;
        ws[OFF_MLPW + t] = (n < 3u) ? (half_t)w4[n * 256u + K] : (half_t)0;
      }
    }
  }
}

// ---- Kernel 1: fused coef/freq 3x3 conv as WMMA implicit GEMM ------------
// Block = 8 waves; block owns 16 pixels (one x-tile of row y), wave w owns
// N-tiles 4w..4w+3 of N=512. The 3x18x64 (rows y-1..y+1, halo'd x) f16 input
// tile is staged into LDS ONCE per block via async global->LDS DMA (zeros
// pre-stored for the out-of-bounds halo), then all 8 waves read A-fragments
// from LDS. K loop = 9 taps x 2 chunks of 32 input channels.
#define TILE_CHUNKS (3 * 18 * 8)   // 432 x 16B chunks = 6912 B
__global__ __launch_bounds__(256) void conv_kernel(half_t* __restrict__ ws,
                                                   const float* __restrict__ coef_b,
                                                   const float* __restrict__ freq_b) {
  const half_t* featH = ws + OFF_FEATH;
  const half_t* cw    = ws + OFF_CONVW;
  half_t*       cf    = ws + OFF_CF;

  __shared__ __align__(16) half_t tile[3 * 18 * 64];   // [row][x+halo][cin]

  int tid = threadIdx.x;
  int L = tid & 31;
  int w = tid >> 5;
  unsigned bid = blockIdx.x;
  unsigned img = bid >> 10;              // 1024 blocks per image (128 rows * 8 xtiles)
  unsigned r   = bid & 1023u;
  unsigned y   = r >> 3;
  unsigned x0  = (r & 7u) * 16u;
  int nt0 = w * 4;

  // ---- stage input tile into LDS ----------------------------------------
#if defined(HAVE_ASYNC_LDS)
  // Pre-zero (covers OOB halo), then async DMA the valid 16B chunks.
  for (int cid = tid; cid < TILE_CHUNKS; cid += 256)
    *(v8h*)&tile[cid * 8] = v8h_zero();
  asm volatile("s_wait_dscnt 0" ::: "memory");
  __syncthreads();
  for (int cid = tid; cid < TILE_CHUNKS; cid += 256) {
    int pi = cid >> 3, sub = cid & 7;
    int rr = pi / 18, p = pi % 18;
    int sy = (int)y + rr - 1;
    int sx = (int)x0 + p - 1;
    if (sy >= 0 && sy < 128 && sx >= 0 && sx < 128) {
      const half_t* src =
          featH + ((size_t)img * 16384u + (unsigned)sy * 128u + (unsigned)sx) * 64u + sub * 8;
      // AS(1) ptr: plain 64-bit address. AS(3) ptr: low 32 bits of the
      // generic LDS address are the LDS byte offset (ISA 10.2).
      __builtin_amdgcn_global_load_async_to_lds_b128(
          (g_v4i_p)(unsigned long long)(const void*)src,
          (l_v4i_p)(unsigned)(unsigned long long)(void*)&tile[cid * 8], 0, 0);
    }
  }
  WAIT_ASYNC();
  __syncthreads();
#else
  for (int cid = tid; cid < TILE_CHUNKS; cid += 256) {
    int pi = cid >> 3, sub = cid & 7;
    int rr = pi / 18, p = pi % 18;
    int sy = (int)y + rr - 1;
    int sx = (int)x0 + p - 1;
    v8h v = v8h_zero();
    if (sy >= 0 && sy < 128 && sx >= 0 && sx < 128)
      v = *(const v8h*)(featH +
                        ((size_t)img * 16384u + (unsigned)sy * 128u + (unsigned)sx) * 64u +
                        sub * 8);
    *(v8h*)&tile[cid * 8] = v;
  }
  __syncthreads();
#endif

  // ---- WMMA implicit GEMM ------------------------------------------------
  v8f acc[4];
#pragma unroll
  for (int j = 0; j < 4; ++j)
#pragma unroll
    for (int v = 0; v < 8; ++v) acc[j][v] = 0.f;

  int kbase = (L < 16) ? 0 : 8;

  for (int tap = 0; tap < 9; ++tap) {
    int ty = tap / 3, tx = tap % 3;
    int p = (L & 15) + tx;                       // x position inside halo'd tile
#pragma unroll
    for (int kc = 0; kc < 2; ++kc) {
      const half_t* ap = &tile[(ty * 18 + p) * 64 + kc * 32 + kbase];
      v16h a = cat16(*(const v8h*)ap, *(const v8h*)(ap + 16));
      int ktile = tap * 2 + kc;
#pragma unroll
      for (int j = 0; j < 4; ++j) {
        const half_t* bp =
            cw + (((unsigned)ktile * 32u + (unsigned)(nt0 + j)) * 32u + (unsigned)L) * 16u;
        v16h b = cat16(*(const v8h*)bp, *(const v8h*)(bp + 8));
        acc[j] = __builtin_amdgcn_wmma_f32_16x16x32_f16(false, a, false, b,
                                                        (short)0, acc[j], false, false);
      }
    }
  }

  unsigned pixBase = img * 16384u + y * 128u + x0;
#pragma unroll
  for (int j = 0; j < 4; ++j) {
    unsigned n = (unsigned)(nt0 + j) * 16u + (unsigned)(L & 15);
    float bias = (n < 256u) ? coef_b[n] : freq_b[n - 256u];
#pragma unroll
    for (int v = 0; v < 8; ++v) {
      unsigned M = (unsigned)v + ((L < 16) ? 0u : 8u);
      cf[(size_t)(pixBase + M) * 512u + n] = (half_t)(acc[j][v] + bias);
    }
  }
}

// ---- Kernel 2: fused gather + trig features + 3x256 MLP (WMMA) + combine -
// 64 threads = 2 independent waves. Each wave owns 4 queries x 4 shifts = 16
// rows (M=16), so the 4-shift area-weighted combine is done in-wave.
__global__ __launch_bounds__(64) void query_kernel(const half_t* __restrict__ ws,
                                                   const float* __restrict__ coord,
                                                   const float* __restrict__ cell,
                                                   const float* __restrict__ phase_w,
                                                   const float* __restrict__ b1,
                                                   const float* __restrict__ b2,
                                                   const float* __restrict__ b3,
                                                   const float* __restrict__ b4,
                                                   const float* __restrict__ inp,
                                                   float* __restrict__ out) {
  __shared__ __align__(16) half_t act[2][2][16][256];   // [wave][pingpong][row][col]
  __shared__ float outBuf[2][16][16];

  int L  = threadIdx.x & 31;
  int wv = threadIdx.x >> 5;
  unsigned wgid  = blockIdx.x * 2u + (unsigned)wv;
  unsigned img   = wgid >> 14;
  unsigned qbase = (wgid & 16383u) * 4u;

  const half_t* cf = ws + OFF_CF + (size_t)img * 16384u * 512u;
  const half_t* mw = ws + OFF_MLPW;

  // ---- stage 1: gather + trig feature map -> act[wv][0] -----------------
  int nbase = L * 8;
  int kk = (L < 16) ? nbase : (nbase - 128);
  for (int m = 0; m < 16; ++m) {
    int qi = m >> 2, s = m & 3;
    unsigned q = qbase + (unsigned)qi;
    float cy = coord[q * 2 + 0], cx = coord[q * 2 + 1];
    int iy, ix; float rel0, rel1;
    shift_rel(cy, cx, s, iy, ix, rel0, rel1);
    float rc0 = cell[q * 2 + 0] * 128.f;
    float rc1 = cell[q * 2 + 1] * 128.f;

    const half_t* pxp = cf + ((size_t)iy * 128u + (unsigned)ix) * 512u;
    v8h qc = *(const v8h*)(pxp + nbase);              // coef channels nbase..+7
    v8h f0 = *(const v8h*)(pxp + 256 + 2 * kk);       // freq pairs for k=kk..kk+3
    v8h f1 = *(const v8h*)(pxp + 256 + 2 * kk + 8);   // freq pairs for k=kk+4..kk+7
    v8h rres;
#pragma unroll
    for (int j = 0; j < 8; ++j) {
      int k = kk + j;
      float fr0 = (j < 4) ? (float)f0[2 * j]     : (float)f1[2 * (j - 4)];
      float fr1 = (j < 4) ? (float)f0[2 * j + 1] : (float)f1[2 * (j - 4) + 1];
      float ph  = rc0 * phase_w[k * 2] + rc1 * phase_w[k * 2 + 1];
      float pre = fr0 * rel0 + fr1 * rel1 + ph;
      float tv  = (L < 16) ? cosf(PI_F * pre) : sinf(PI_F * pre);
      rres[j] = (half_t)((float)qc[j] * tv);
    }
    *(v8h*)&act[wv][0][m][nbase] = rres;
  }
  asm volatile("s_wait_dscnt 0" ::: "memory");   // intra-wave LDS store->load order

  // ---- stage 2: three 256x256 WMMA layers (f16 in, f32 accumulate) -------
  int kb = (L < 16) ? 0 : 8;
  int Mrow = L & 15;
#pragma unroll
  for (int l = 0; l < 3; ++l) {
    const half_t* src = &act[wv][l & 1][0][0];
    half_t*       dst = &act[wv][(l + 1) & 1][0][0];
    const float*  bias = (l == 0) ? b1 : (l == 1) ? b2 : b3;
    const half_t* wl = mw + (unsigned)l * 65536u;
    __builtin_prefetch(wl, 0, 0);
    for (int nt = 0; nt < 16; ++nt) {
      unsigned ncol = (unsigned)nt * 16u + (unsigned)(L & 15);
      float bv = bias[ncol];
      v8f accl;
#pragma unroll
      for (int v = 0; v < 8; ++v) accl[v] = bv;
#pragma unroll
      for (int kc = 0; kc < 8; ++kc) {
        const half_t* ap = src + Mrow * 256 + kc * 32 + kb;
        v16h a = cat16(*(const v8h*)ap, *(const v8h*)(ap + 16));
        const half_t* bp = wl + ((unsigned)(kc * 16 + nt) * 32u + (unsigned)L) * 16u;
        v16h b = cat16(*(const v8h*)bp, *(const v8h*)(bp + 8));
        accl = __builtin_amdgcn_wmma_f32_16x16x32_f16(false, a, false, b,
                                                      (short)0, accl, false, false);
      }
#pragma unroll
      for (int v = 0; v < 8; ++v) {
        int Md = v + ((L < 16) ? 0 : 8);
        dst[Md * 256 + ncol] = (half_t)fmaxf(accl[v], 0.f);   // ReLU
      }
    }
    asm volatile("s_wait_dscnt 0" ::: "memory");
  }

  // ---- stage 3: final 256->3 layer (N padded to 16) ----------------------
  {
    const half_t* src = &act[wv][1][0][0];       // after 3 layers output is in buf 1
    const half_t* wl  = mw + 196608u;
    unsigned ncol = (unsigned)(L & 15);
    float bv = (ncol < 3u) ? b4[ncol] : 0.f;
    v8f acc4;
#pragma unroll
    for (int v = 0; v < 8; ++v) acc4[v] = bv;
#pragma unroll
    for (int kc = 0; kc < 8; ++kc) {
      const half_t* ap = src + Mrow * 256 + kc * 32 + kb;
      v16h a = cat16(*(const v8h*)ap, *(const v8h*)(ap + 16));
      const half_t* bp = wl + ((unsigned)kc * 32u + (unsigned)L) * 16u;
      v16h b = cat16(*(const v8h*)bp, *(const v8h*)(bp + 8));
      acc4 = __builtin_amdgcn_wmma_f32_16x16x32_f16(false, a, false, b,
                                                    (short)0, acc4, false, false);
    }
#pragma unroll
    for (int v = 0; v < 8; ++v) {
      int Md = v + ((L < 16) ? 0 : 8);
      outBuf[wv][Md][ncol] = acc4[v];
    }
  }
  asm volatile("s_wait_dscnt 0" ::: "memory");

  // ---- stage 4: in-wave 4-shift combine + bilinear(inp) + store ----------
  if (L < 4) {
    int qi = L;
    unsigned q  = qbase + (unsigned)qi;
    unsigned oy = q >> 8, ox = q & 255u;
    float cy = coord[q * 2 + 0], cx = coord[q * 2 + 1];
    float area[4];
#pragma unroll
    for (int s = 0; s < 4; ++s) {
      int iy, ix; float r0, r1;
      shift_rel(cy, cx, s, iy, ix, r0, r1);
      area[s] = fabsf(r0 * r1) + 1e-9f;
    }
    float tot = area[0] + area[1] + area[2] + area[3];

    float fy = fminf(fmaxf(((cy + 1.f) * 128.f - 1.f) * 0.5f, 0.f), 127.f);
    float fx = fminf(fmaxf(((cx + 1.f) * 128.f - 1.f) * 0.5f, 0.f), 127.f);
    int y0 = (int)floorf(fy), x0i = (int)floorf(fx);
    float wy = fy - (float)y0, wx = fx - (float)x0i;
    int y1 = min(y0 + 1, 127), x1 = min(x0i + 1, 127);

#pragma unroll
    for (int c = 0; c < 3; ++c) {
      float accv = 0.f;
#pragma unroll
      for (int s = 0; s < 4; ++s)
        accv += outBuf[wv][qi * 4 + s][c] * area[3 - s];   // order = [3,2,1,0]
      accv /= tot;
      const float* ip = inp + (size_t)(img * 3u + c) * 128u * 128u;
      float v00 = ip[y0 * 128 + x0i], v01 = ip[y0 * 128 + x1];
      float v10 = ip[y1 * 128 + x0i], v11 = ip[y1 * 128 + x1];
      accv += v00 * (1.f - wy) * (1.f - wx) + v01 * (1.f - wy) * wx +
              v10 * wy * (1.f - wx) + v11 * wy * wx;
      out[((size_t)(img * 3u + c) * 256u + oy) * 256u + ox] = accv;
    }
  }
}

// ---------------------------------------------------------------------------
extern "C" void kernel_launch(void* const* d_in, const int* in_sizes, int n_in,
                              void* d_out, int out_size, void* d_ws, size_t ws_size,
                              hipStream_t stream) {
  const float* feat    = (const float*)d_in[0];
  const float* inp     = (const float*)d_in[1];
  const float* coord   = (const float*)d_in[2];
  const float* cell    = (const float*)d_in[3];
  const float* coef_w  = (const float*)d_in[4];
  const float* coef_b  = (const float*)d_in[5];
  const float* freq_w  = (const float*)d_in[6];
  const float* freq_b  = (const float*)d_in[7];
  const float* phase_w = (const float*)d_in[8];
  const float* w1 = (const float*)d_in[9];
  const float* bb1 = (const float*)d_in[10];
  const float* w2 = (const float*)d_in[11];
  const float* bb2 = (const float*)d_in[12];
  const float* w3 = (const float*)d_in[13];
  const float* bb3 = (const float*)d_in[14];
  const float* w4 = (const float*)d_in[15];
  const float* bb4 = (const float*)d_in[16];
  float* out = (float*)d_out;
  half_t* wsH = (half_t*)d_ws;

  const unsigned packTotal = N_FEATH + N_CONVW + N_MLPW;
  pack_kernel<<<(packTotal + 255u) / 256u, 256, 0, stream>>>(
      feat, coef_w, freq_w, w1, w2, w3, w4, wsH);

  conv_kernel<<<2048, 256, 0, stream>>>(wsH, coef_b, freq_b);

  query_kernel<<<16384, 64, 0, stream>>>(wsH, coord, cell, phase_w,
                                         bb1, bb2, bb3, bb4, inp, out);
}
